// PaddedSelectEntityActionHead_69475390980244
// MI455X (gfx1250) — matrix operand order repaired and checked by reading, not compiled, not run on hardware
//
#include <hip/hip_runtime.h>

typedef __attribute__((ext_vector_type(16))) _Float16 v16h;
typedef __attribute__((ext_vector_type(8)))  _Float16 v8h;
typedef __attribute__((ext_vector_type(8)))  float    v8f;

#define B_ENV 512
#define MA    64
#define ME    256
#define DM    256
#define DQK   32
#define SCALE 0.17677669529663687f   /* 1/sqrt(32) */

// ---------------- workspace layout (bytes) ----------------
// pq_h   : B_ENV*MA*DQK * 2   = 2,097,152
// pk_h   : B_ENV*ME*DQK * 2   = 8,388,608
// qmask  : B_ENV      * 8     =     4,096   (1 bit / q slot, 64 slots/env)
// kmask  : B_ENV * 4  * 8     =    16,384   (1 bit / k slot, 256 slots/env)
#define OFF_PQ 0
#define OFF_PK (2097152)
#define OFF_QM (2097152 + 8388608)
#define OFF_KM (2097152 + 8388608 + 4096)

// ---------------------------------------------------------------------------
// Zero validity bitmasks (every call: deterministic, no stale state).
// qmask: 512 u64, kmask: 2048 u64 -> 2560 u64 total.
// ---------------------------------------------------------------------------
__global__ void k_init(unsigned long long* __restrict__ qmask,
                       unsigned long long* __restrict__ kmask) {
  int i = blockIdx.x * blockDim.x + threadIdx.x;
  if (i < B_ENV) qmask[i] = 0ull;
  if (i < B_ENV * 4) kmask[i] = 0ull;
}

// ---------------------------------------------------------------------------
// Ragged gather + [256 -> 32] projection, scatter to padded f16 buffer.
// One wave per entity; lane j owns output dim j. W (256x32 f32) lives in LDS.
// Validity recorded as one bit per padded slot: mask[dst>>6] |= 1<<(dst&63)
// (identical formula for q and k because MA*B and ME*B are both slot counts).
// ---------------------------------------------------------------------------
__global__ void k_project(const float* __restrict__ x,
                          const float* __restrict__ W,
                          const float* __restrict__ bias,
                          const int*   __restrict__ rows,
                          const int*   __restrict__ scatter,
                          int count,
                          _Float16* __restrict__ outH,
                          unsigned long long* __restrict__ mask) {
  __shared__ float lW[DM * DQK];                 // 32 KB
  for (int i = threadIdx.x; i < DM * DQK; i += blockDim.x) lW[i] = W[i];
  __syncthreads();

  const int lane = threadIdx.x & 31;
  const int wid  = threadIdx.x >> 5;
  const int wpb  = blockDim.x >> 5;
  const int gw   = blockIdx.x * wpb + wid;
  const int nw   = gridDim.x * wpb;

  for (int a = gw; a < count; a += nw) {
    const int row = rows[a];                     // wave-uniform gather index
    const float* xr = x + (size_t)row * DM;
    float acc = bias[lane];
#pragma unroll 8
    for (int i = 0; i < DM; ++i)
      acc = fmaf(xr[i], lW[i * DQK + lane], acc);
    const int dst = scatter[a];
    outH[(size_t)dst * DQK + lane] = (_Float16)acc;
    if (lane == 0)
      atomicOr(&mask[dst >> 6], 1ull << (dst & 63));
  }
}

// ---------------------------------------------------------------------------
// Batched logits: one wave computes a 16x64 strip = 4 x v_wmma_f32_16x16x32_f16
// (K = d_qk = 32), reusing the A fragment across the 4 k-tiles. The 4 k-tiles'
// column-validity bits all live in ONE wave-uniform u64 of kmask; the 16 row
// bits live in two bytes (one per lane half) of the wave-uniform qmask word -> no LDS, no bpermute.
// ---------------------------------------------------------------------------
__global__ void k_logits(const _Float16* __restrict__ pq,
                         const _Float16* __restrict__ pk,
                         const unsigned long long* __restrict__ qmask,
                         const unsigned long long* __restrict__ kmask,
                         float* __restrict__ outLogits) {
  const int lane = threadIdx.x & 31;
  const int wid  = threadIdx.x >> 5;
  const int task = blockIdx.x * (blockDim.x >> 5) + wid;   // 0 .. 8191
  const int env  = task >> 4;                              // task / 16
  const int t    = task & 15;
  const int qt   = t >> 2;                                 // 0..3  (MA/16)
  const int ktg  = t & 3;                                  // 0..3  (4 k-tiles)
  const int l15  = lane & 15;
  const int hi   = lane >> 4;                              // 0 or 1

  // ---- A fragment: 16-bit A-matrix 16x32 layout ----
  // lane L: row M = L&15; halves {b..b+7} and {b+16..b+23}, b = (L<16)?0:8
  const v8h* qp = (const v8h*)(pq + ((size_t)env * MA + qt * 16 + l15) * DQK);
  v8h alo = qp[hi];        // halves 0-7  (or 8-15)
  v8h ahi = qp[hi + 2];    // halves 16-23 (or 24-31)
  v16h afrag = __builtin_shufflevector(alo, ahi,
      0, 1, 2, 3, 4, 5, 6, 7, 8, 9, 10, 11, 12, 13, 14, 15);

  // ---- validity: wave-uniform mask words, per-lane bit tests ----
  const unsigned long long qw = qmask[env];
  const unsigned int qbyte =
      (unsigned int)(qw >> (qt * 16 + hi * 8)) & 0xffu;    // this lane's 8 rows
  const unsigned long long kw = kmask[env * 4 + ktg];      // 64 columns

  const int mbase = qt * 16 + hi * 8;

#pragma unroll
  for (int kk = 0; kk < 4; ++kk) {
    const int kt = ktg * 4 + kk;

    // ---- B fragment: 16-bit B-matrix 32x16 layout ----
    // lane L: column N = L&15; K halves {b..b+7} and {b+16..b+23}, b = (L<16)?0:8
    const v8h* kp = (const v8h*)(pk + ((size_t)env * ME + kt * 16 + l15) * DQK);
    v8h blo = kp[hi];
    v8h bhi = kp[hi + 2];
    v16h bfrag = __builtin_shufflevector(blo, bhi,
        0, 1, 2, 3, 4, 5, 6, 7, 8, 9, 10, 11, 12, 13, 14, 15);

    v8f c = {};
    v8f d = __builtin_amdgcn_wmma_f32_16x16x32_f16(
        /*neg_a=*/false, afrag, /*neg_b=*/false, bfrag,
        /*c_mod=*/(short)0, c, /*reuse_a=*/false, /*reuse_b=*/false);

    const unsigned int kok = (unsigned int)(kw >> (kk * 16 + l15)) & 1u;

    // D layout: lane L, VGPR r -> M = r + (L<16?0:8), N = L&15
    const size_t rowbase = (size_t)env * MA * ME + (size_t)kt * 16 + l15;
#pragma unroll
    for (int r = 0; r < 8; ++r) {
      const int m = mbase + r;                             // 0..63 within env
      const unsigned int qok = (qbyte >> r) & 1u;
      const float val = (qok & kok) ? d[r] * SCALE : -1e9f;
      outLogits[rowbase + (size_t)m * ME] = val;
    }
  }
}

// ---------------------------------------------------------------------------
// Per-actor log-softmax over 256 logits: logprob at prev_action + entropy.
// One wave per actor; shfl_xor reductions (wave32).
// ---------------------------------------------------------------------------
__global__ void k_head(const float* __restrict__ logits,
                       const int* __restrict__ qindices,
                       const int* __restrict__ prev_actions,
                       int A,
                       float* __restrict__ outAct,
                       float* __restrict__ outLp,
                       float* __restrict__ outEnt) {
  const int lane = threadIdx.x & 31;
  const int a = blockIdx.x * (blockDim.x >> 5) + (threadIdx.x >> 5);
  if (a >= A) return;

  const int dest = qindices[a];
  const float* row = logits + (size_t)dest * ME;

  float v[8];
#pragma unroll
  for (int c = 0; c < 8; ++c) v[c] = row[lane + 32 * c];

  float m = v[0];
#pragma unroll
  for (int c = 1; c < 8; ++c) m = fmaxf(m, v[c]);
  for (int off = 16; off; off >>= 1) m = fmaxf(m, __shfl_xor(m, off, 32));

  float s = 0.f;
#pragma unroll
  for (int c = 0; c < 8; ++c) s += __expf(v[c] - m);
  for (int off = 16; off; off >>= 1) s += __shfl_xor(s, off, 32);

  const float lse = m + __logf(s);

  float ent = 0.f;
#pragma unroll
  for (int c = 0; c < 8; ++c) {
    const float t = v[c] - lse;
    ent += __expf(t) * t;                      // exp underflows to 0 for -1e9
  }
  for (int off = 16; off; off >>= 1) ent += __shfl_xor(ent, off, 32);

  if (lane == 0) {
    const int act = prev_actions[a];
    outAct[a] = (float)act;
    outLp[a]  = row[act] - lse;
    outEnt[a] = -ent;
  }
}

// ---------------------------------------------------------------------------
extern "C" void kernel_launch(void* const* d_in, const int* in_sizes, int n_in,
                              void* d_out, int out_size, void* d_ws, size_t ws_size,
                              hipStream_t stream) {
  const float* x            = (const float*)d_in[0];
  const float* Wq           = (const float*)d_in[1];
  const float* bq           = (const float*)d_in[2];
  const float* Wk           = (const float*)d_in[3];
  const float* bk           = (const float*)d_in[4];
  const int*   actors       = (const int*)d_in[5];
  const int*   qindices     = (const int*)d_in[6];
  const int*   actees       = (const int*)d_in[7];
  const int*   kindices     = (const int*)d_in[8];
  const int*   prev_actions = (const int*)d_in[9];

  const int A = in_sizes[5];   // #actors  (== in_sizes[6])
  const int E = in_sizes[7];   // #actees  (== in_sizes[8])

  char* ws = (char*)d_ws;
  _Float16* pq = (_Float16*)(ws + OFF_PQ);
  _Float16* pk = (_Float16*)(ws + OFF_PK);
  unsigned long long* qmask = (unsigned long long*)(ws + OFF_QM);
  unsigned long long* kmask = (unsigned long long*)(ws + OFF_KM);

  float* out       = (float*)d_out;
  float* outAct    = out;
  float* outLp     = out + A;
  float* outEnt    = out + 2 * (size_t)A;
  float* outLogits = out + 3 * (size_t)A;

  // 1) clear validity bitmasks (2560 u64)
  k_init<<<(B_ENV * 4 + 255) / 256, 256, 0, stream>>>(qmask, kmask);
  // 2) project + scatter queries and keys (f16 staging)
  k_project<<<512, 256, 0, stream>>>(x, Wq, bq, actors, qindices, A, pq, qmask);
  k_project<<<512, 256, 0, stream>>>(x, Wk, bk, actees, kindices, E, pk, kmask);
  // 3) WMMA batched logits: 512 envs * 4 qtiles * 4 ktile-groups, 8 waves/block
  k_logits<<<(B_ENV * 4 * 4) / 8, 256, 0, stream>>>(pq, pk, qmask, kmask, outLogits);
  // 4) per-actor softmax head, 8 actors per block
  k_head<<<(A + 7) / 8, 256, 0, stream>>>(outLogits, qindices, prev_actions, A,
                                          outAct, outLp, outEnt);
}
